// SimpleLstmActor_11957188952414
// MI455X (gfx1250) — compile-verified, matrix operand order checked by reading, not run on hardware
//
#include <hip/hip_runtime.h>
#include <hip/hip_bf16.h>

// ---------------------------------------------------------------------------
// SimpleLstmActor for MI455X (gfx1250): bf16 WMMA everywhere.
//   T=256, B=2048, OBS=512, MLP 512->128->128->64, LSTM(64,32), head 32->16
// ---------------------------------------------------------------------------

#define T_STEPS 256
#define BATCH   2048
#define OBS     512
#define NACT    16
#define HID     32
#define TB      (T_STEPS * BATCH)

// LDS row pitches (bf16 elements), padded +8 (16B) so that the 16 fragment
// rows land on disjoint 4-bank windows of the 64 LDS banks (no conflicts),
// while keeping 16B alignment for ds_load_b128 merging.
#define P1 (OBS + 8)      // 520 : W1 rows
#define P2 (128 + 8)      // 136 : W2/W3/activation rows
#define PG (96 + 8)       // 104 : LSTM gate-weight / A-buffer rows
#define PL (HID + 8)      // 40  : Wlog rows

typedef __attribute__((ext_vector_type(16))) __bf16       v16bf;
typedef __attribute__((ext_vector_type(8)))  float        v8f;
typedef __attribute__((ext_vector_type(8)))  unsigned int v8u;

static __device__ __forceinline__ v8f wmma_bf16(v16bf a, v16bf b, v8f c) {
  return __builtin_amdgcn_wmma_f32_16x16x32_bf16(false, a, false, b, (short)0, c,
                                                 false, false);
}

// two f32 -> packed bf16 pair: +0x8000 round-half-up, then one v_perm_b32
static __device__ __forceinline__ unsigned int pkbf(float lo, float hi) {
  const unsigned int a = __builtin_bit_cast(unsigned int, lo) + 0x8000u;
  const unsigned int b = __builtin_bit_cast(unsigned int, hi) + 0x8000u;
  return __builtin_amdgcn_perm(b, a, 0x07060302u);  // {b.b3,b.b2,a.b3,a.b2}
}
static __device__ __forceinline__ __bf16 bf16_rh(float x) {
  const unsigned int u = (__builtin_bit_cast(unsigned int, x) + 0x8000u) >> 16;
  return __builtin_bit_cast(__bf16, (unsigned short)u);
}

// fast transcendentals: v_exp_f32 / v_rcp_f32 based
static __device__ __forceinline__ float tanh_fast(float x) {
  const float ax = __builtin_fabsf(x);
  const float e  = __builtin_amdgcn_exp2f(ax * 2.8853900817779268f); // e^(2|x|)
  const float r  = __builtin_amdgcn_rcpf(e + 1.f);
  return __builtin_copysignf(1.f - 2.f * r, x);
}
static __device__ __forceinline__ float sigmoid_fast(float x) {
  const float e = __builtin_amdgcn_exp2f(x * -1.4426950408889634f);
  return __builtin_amdgcn_rcpf(1.f + e);
}

// 16-bit operand fragment, CDNA5 layout (ISA 7.12.2, 16-bit A 16x32):
//   row = lane&15 ; K-halves split by lane>=16 (+8)
//   VGPR p<4 : K = 2p+{0,1}(+khi) ; p>=4 : K = 16+2(p-4)+{0,1}(+khi)
// B operand assumed symmetric with N = lane&15 (row-major [out][in] weights).
static __device__ __forceinline__ v16bf frag_lds(const __bf16* __restrict__ base,
                                                 int ld, int koff) {
  const int lane = threadIdx.x & 31;
  const int row  = lane & 15;
  const int khi  = (lane >> 4) << 3;
  v16bf f;
#pragma unroll
  for (int p = 0; p < 8; ++p) {
    const int kb = ((p & 3) << 1) + ((p >> 2) << 4) + khi;
    const __bf16* s = base + row * ld + koff + kb;
    f[2 * p]     = s[0];
    f[2 * p + 1] = s[1];
  }
  return f;
}

// Same fragment from row-major f32 global memory: 4x global_load_b128 + 8 perms.
static __device__ __forceinline__ v16bf fragA_gf32(const float* __restrict__ base,
                                                   int ld, int koff) {
  const int lane = threadIdx.x & 31;
  const int row  = lane & 15;
  const int khi  = (lane >> 4) << 3;
  const float* p = base + row * ld + koff + khi;
  const float4 v0 = *reinterpret_cast<const float4*>(p);
  const float4 v1 = *reinterpret_cast<const float4*>(p + 4);
  const float4 v2 = *reinterpret_cast<const float4*>(p + 16);
  const float4 v3 = *reinterpret_cast<const float4*>(p + 20);
  v8u u;
  u[0] = pkbf(v0.x, v0.y); u[1] = pkbf(v0.z, v0.w);
  u[2] = pkbf(v1.x, v1.y); u[3] = pkbf(v1.z, v1.w);
  u[4] = pkbf(v2.x, v2.y); u[5] = pkbf(v2.z, v2.w);
  u[6] = pkbf(v3.x, v3.y); u[7] = pkbf(v3.z, v3.w);
  return __builtin_bit_cast(v16bf, u);
}

// ---------------------------------------------------------------------------
// Kernel 0: weight prep (f32 -> bf16; fuse LSTM weights/biases)
// ---------------------------------------------------------------------------
__global__ void prep_kernel(const float* __restrict__ W1, const float* __restrict__ W2,
                            const float* __restrict__ W3, const float* __restrict__ Wih,
                            const float* __restrict__ Whh, const float* __restrict__ bih,
                            const float* __restrict__ bhh, const float* __restrict__ Wlog,
                            __bf16* __restrict__ W1b, __bf16* __restrict__ W2b,
                            __bf16* __restrict__ W3b, __bf16* __restrict__ Wgb,
                            float* __restrict__ biasg, __bf16* __restrict__ Wlogb) {
  const int i = blockIdx.x * blockDim.x + threadIdx.x;
  if (i < 128 * OBS) W1b[i] = (__bf16)W1[i];
  if (i < 128 * 128) W2b[i] = (__bf16)W2[i];
  if (i < 64 * 128)  W3b[i] = (__bf16)W3[i];
  if (i < 128 * 96) {               // [Wih | Whh] -> [128][96]
    const int n = i / 96, k = i % 96;
    const float v = (k < 64) ? Wih[n * 64 + k] : Whh[n * HID + (k - 64)];
    Wgb[i] = (__bf16)v;
  }
  if (i < 128) biasg[i] = bih[i] + bhh[i];
  if (i < NACT * HID) Wlogb[i] = (__bf16)Wlog[i];
}

// ---------------------------------------------------------------------------
// Kernel 1: fused MLP encoder.  W1/W2/W3 (bf16, bank-conflict-padded) +
// 16 waves of activation scratch = 324,608 B dynamic LDS -> 1 WG/WGP.
// Each wave: one 16-row tile, 176 WMMAs.  z written as bf16.
// ---------------------------------------------------------------------------
__global__ void __launch_bounds__(512) mlp_encoder_kernel(
    const float* __restrict__ x,  const float* __restrict__ b1,
    const float* __restrict__ b2, const float* __restrict__ b3,
    const __bf16* __restrict__ gW1, const __bf16* __restrict__ gW2,
    const __bf16* __restrict__ gW3, __bf16* __restrict__ z) {
  extern __shared__ __bf16 smem[];
  __bf16* sW1  = smem;                  // 128 x P1
  __bf16* sW2  = sW1 + 128 * P1;        // 128 x P2
  __bf16* sW3  = sW2 + 128 * P2;        // 64  x P2
  __bf16* sScr = sW3 + 64 * P2;         // per-wave 2 x 16 x P2

  for (int i = threadIdx.x; i < 128 * OBS; i += blockDim.x)
    sW1[(i >> 9) * P1 + (i & (OBS - 1))] = gW1[i];
  for (int i = threadIdx.x; i < 128 * 128; i += blockDim.x)
    sW2[(i >> 7) * P2 + (i & 127)] = gW2[i];
  for (int i = threadIdx.x; i < 64 * 128; i += blockDim.x)
    sW3[(i >> 7) * P2 + (i & 127)] = gW3[i];
  __syncthreads();

  const int lane = threadIdx.x & 31;
  const int wave = threadIdx.x >> 5;
  const int ncol = lane & 15;      // C-fragment: N = lane&15
  const int hi   = lane >> 4;      // C-fragment: M = r + 8*hi
  __bf16* z1buf = sScr + wave * (2 * 16 * P2);
  __bf16* z2buf = z1buf + 16 * P2;

  const int gwave = (blockIdx.x * blockDim.x + threadIdx.x) >> 5;
  const int nwave = (gridDim.x * blockDim.x) >> 5;

  for (int tile = gwave; tile < TB / 16; tile += nwave) {
    const float* xt = x + (size_t)tile * 16 * OBS;

    // prefetch next tile (global_prefetch_b8): 256 cachelines of 128B
    const int tnext = tile + nwave;
    if (tnext < TB / 16) {
      const float* xn = x + (size_t)tnext * 16 * OBS;
#pragma unroll
      for (int i = 0; i < 8; ++i)
        __builtin_prefetch(xn + (size_t)(lane + i * 32) * 32, 0, 3);
    }

    // ---- layer 1: [16x512] x [512x128] ----
    v8f acc[8];
#pragma unroll
    for (int nt = 0; nt < 8; ++nt)
#pragma unroll
      for (int r = 0; r < 8; ++r) acc[nt][r] = 0.f;
    for (int ks = 0; ks < 16; ++ks) {
      const v16bf a = fragA_gf32(xt, OBS, ks * 32);
#pragma unroll
      for (int nt = 0; nt < 8; ++nt) {
        const v16bf b = frag_lds(sW1 + nt * 16 * P1, P1, ks * 32);
        acc[nt] = wmma_bf16(a, b, acc[nt]);
      }
    }
#pragma unroll
    for (int nt = 0; nt < 8; ++nt) {
      const float bias = b1[nt * 16 + ncol];
#pragma unroll
      for (int r = 0; r < 8; ++r)
        z1buf[(r + hi * 8) * P2 + nt * 16 + ncol] = bf16_rh(tanh_fast(acc[nt][r] + bias));
    }

    // ---- layer 2: [16x128] x [128x128] ----
#pragma unroll
    for (int nt = 0; nt < 8; ++nt)
#pragma unroll
      for (int r = 0; r < 8; ++r) acc[nt][r] = 0.f;
#pragma unroll
    for (int ks = 0; ks < 4; ++ks) {
      const v16bf a = frag_lds(z1buf, P2, ks * 32);
#pragma unroll
      for (int nt = 0; nt < 8; ++nt) {
        const v16bf b = frag_lds(sW2 + nt * 16 * P2, P2, ks * 32);
        acc[nt] = wmma_bf16(a, b, acc[nt]);
      }
    }
#pragma unroll
    for (int nt = 0; nt < 8; ++nt) {
      const float bias = b2[nt * 16 + ncol];
#pragma unroll
      for (int r = 0; r < 8; ++r)
        z2buf[(r + hi * 8) * P2 + nt * 16 + ncol] = bf16_rh(tanh_fast(acc[nt][r] + bias));
    }

    // ---- layer 3: [16x128] x [128x64] (linear), store z as bf16 ----
    v8f acc3[4];
#pragma unroll
    for (int nt = 0; nt < 4; ++nt)
#pragma unroll
      for (int r = 0; r < 8; ++r) acc3[nt][r] = 0.f;
#pragma unroll
    for (int ks = 0; ks < 4; ++ks) {
      const v16bf a = frag_lds(z2buf, P2, ks * 32);
#pragma unroll
      for (int nt = 0; nt < 4; ++nt) {
        const v16bf b = frag_lds(sW3 + nt * 16 * P2, P2, ks * 32);
        acc3[nt] = wmma_bf16(a, b, acc3[nt]);
      }
    }
#pragma unroll
    for (int nt = 0; nt < 4; ++nt) {
      const float bias = b3[nt * 16 + ncol];
#pragma unroll
      for (int r = 0; r < 8; ++r)
        z[(size_t)(tile * 16 + r + hi * 8) * 64 + nt * 16 + ncol] =
            bf16_rh(acc3[nt][r] + bias);
    }
  }
}

// ---------------------------------------------------------------------------
// Kernel 2: LSTM scan + logits head, fused.  128 waves each own a 16-row
// batch tile for all 256 steps.  All 25 loop-invariant B fragments (gates +
// logits) and biases are hoisted into VGPRs before the time loop (gfx1250
// VGPR-MSB allows >256 VGPRs; 1 wave/SIMD so no occupancy cost).  h,c stay
// in registers mapped to the C-fragment layout.
// ---------------------------------------------------------------------------
__global__ void __launch_bounds__(128) lstm_head_kernel(
    const __bf16* __restrict__ z, const int* __restrict__ done,
    const int* __restrict__ action,
    const __bf16* __restrict__ gWg, const float* __restrict__ gBiasG,
    const __bf16* __restrict__ gWlog, const float* __restrict__ blog,
    const float* __restrict__ h0, const float* __restrict__ c0,
    float* __restrict__ outLogp, float* __restrict__ outEnt) {
  __shared__ __bf16 sWg[128 * PG];       // [gate n][k]  k = [z(64) | h(32)]
  __shared__ __bf16 sWlog[NACT * PL];
  __shared__ float  sBias[128];
  __shared__ float  sBlog[NACT];
  __shared__ __bf16 sA[4][16 * PG];      // per-wave A buffer [row][z|h]

  for (int i = threadIdx.x; i < 128 * 96; i += blockDim.x)
    sWg[(i / 96) * PG + (i % 96)] = gWg[i];
  for (int i = threadIdx.x; i < NACT * HID; i += blockDim.x)
    sWlog[(i >> 5) * PL + (i & 31)] = gWlog[i];
  for (int i = threadIdx.x; i < 128; i += blockDim.x) sBias[i] = gBiasG[i];
  for (int i = threadIdx.x; i < NACT; i += blockDim.x) sBlog[i] = blog[i];
  __syncthreads();

  const int lane = threadIdx.x & 31;
  const int wave = threadIdx.x >> 5;
  const int ncol = lane & 15;
  const int hi   = lane >> 4;
  __bf16* abuf = &sA[wave][0];

  const int tile    = blockIdx.x * 4 + wave;   // 0..127
  const int rowbase = tile * 16;

  // hoist all loop-invariant operands into registers
  v16bf bW[3][8];
#pragma unroll
  for (int ks = 0; ks < 3; ++ks)
#pragma unroll
    for (int nt = 0; nt < 8; ++nt)
      bW[ks][nt] = frag_lds(sWg + nt * 16 * PG, PG, ks * 32);
  const v16bf bLg = frag_lds(sWlog, PL, 0);
  float bs[8];
#pragma unroll
  for (int nt = 0; nt < 8; ++nt) bs[nt] = sBias[nt * 16 + ncol];
  const float blg = sBlog[ncol];

  float hreg[16], creg[16];
#pragma unroll
  for (int r = 0; r < 8; ++r)
#pragma unroll
    for (int hf = 0; hf < 2; ++hf) {
      const int row = rowbase + r + hi * 8;
      const int j   = hf * 16 + ncol;
      hreg[r * 2 + hf] = h0[row * HID + j];
      creg[r * 2 + hf] = c0[row * HID + j];
    }

  const float L2E = 1.4426950408889634f;
  const float LN2 = 0.6931471805599453f;

  for (int t = 0; t < T_STEPS; ++t) {
    // z_t tile (bf16) -> abuf cols 0..63 : pure b128 copy, 1 cacheline / 8 lanes
    const __bf16* zt = z + ((size_t)t * BATCH + rowbase) * 64;
#pragma unroll
    for (int i = 0; i < 4; ++i) {
      const int idx = lane + i * 32;
      const int rr  = idx >> 3;
      const int cc  = (idx & 7) << 3;
      *reinterpret_cast<uint4*>(abuf + rr * PG + cc) =
          *reinterpret_cast<const uint4*>(zt + rr * 64 + cc);
    }
    // done-mask state; masked h -> abuf cols 64..95
#pragma unroll
    for (int r = 0; r < 8; ++r) {
      const int row  = rowbase + r + hi * 8;
      const float mk = 1.f - (float)done[t * BATCH + row];
#pragma unroll
      for (int hf = 0; hf < 2; ++hf) {
        hreg[r * 2 + hf] *= mk;
        creg[r * 2 + hf] *= mk;
        abuf[(r + hi * 8) * PG + 64 + hf * 16 + ncol] = bf16_rh(hreg[r * 2 + hf]);
      }
    }
    // gates = [z|h] @ [Wih|Whh]^T + bias : 8 N-tiles x 3 K-slices
    v8f acc[8];
#pragma unroll
    for (int nt = 0; nt < 8; ++nt)
#pragma unroll
      for (int r = 0; r < 8; ++r) acc[nt][r] = bs[nt];
#pragma unroll
    for (int ks = 0; ks < 3; ++ks) {
      const v16bf a = frag_lds(abuf, PG, ks * 32);
#pragma unroll
      for (int nt = 0; nt < 8; ++nt)
        acc[nt] = wmma_bf16(a, bW[ks][nt], acc[nt]);
    }
    // pointwise LSTM cell (torch gate order i,f,g,o = tiles {0,1},{2,3},{4,5},{6,7})
#pragma unroll
    for (int r = 0; r < 8; ++r)
#pragma unroll
      for (int hf = 0; hf < 2; ++hf) {
        const float ig = acc[0 + hf][r], fg = acc[2 + hf][r];
        const float gg = acc[4 + hf][r], og = acc[6 + hf][r];
        float cc = sigmoid_fast(fg) * creg[r * 2 + hf] + sigmoid_fast(ig) * tanh_fast(gg);
        float hh = sigmoid_fast(og) * tanh_fast(cc);
        creg[r * 2 + hf] = cc;
        hreg[r * 2 + hf] = hh;
        abuf[(r + hi * 8) * PG + 64 + hf * 16 + ncol] = bf16_rh(hh);
      }
    // logits = h @ Wlog^T + blog : single 16x16x32 WMMA
    v8f lg;
#pragma unroll
    for (int r = 0; r < 8; ++r) lg[r] = blg;
    {
      const v16bf a = frag_lds(abuf, PG, 64);   // K-slice = h columns
      lg = wmma_bf16(a, bLg, lg);
    }
    // log-softmax / gather / entropy: row r's 16 logits live in one 16-lane group
#pragma unroll
    for (int r = 0; r < 8; ++r) {
      const float v = lg[r];
      float mx = v;
#pragma unroll
      for (int off = 8; off > 0; off >>= 1) mx = fmaxf(mx, __shfl_xor(mx, off, 16));
      float s = __builtin_amdgcn_exp2f((v - mx) * L2E);
#pragma unroll
      for (int off = 8; off > 0; off >>= 1) s += __shfl_xor(s, off, 16);
      const float lp = v - mx - __builtin_amdgcn_logf(s) * LN2;
      float se = lp * __builtin_amdgcn_exp2f(lp * L2E);
#pragma unroll
      for (int off = 8; off > 0; off >>= 1) se += __shfl_xor(se, off, 16);
      const int row = rowbase + r + hi * 8;
      const int a_t = action[t * BATCH + row];
      if (ncol == a_t) outLogp[t * BATCH + row] = lp;
      if (ncol == 0)   outEnt[t * BATCH + row] = -se;
    }
  }
}

// ---------------------------------------------------------------------------
// Kernel 3: action pass-through (output slot 0)
// ---------------------------------------------------------------------------
__global__ void action_copy_kernel(const int* __restrict__ action,
                                   float* __restrict__ out) {
  const int i = blockIdx.x * blockDim.x + threadIdx.x;
  if (i < TB) out[i] = (float)action[i];
}

// ---------------------------------------------------------------------------
extern "C" void kernel_launch(void* const* d_in, const int* in_sizes, int n_in,
                              void* d_out, int out_size, void* d_ws, size_t ws_size,
                              hipStream_t stream) {
  (void)in_sizes; (void)n_in; (void)out_size; (void)ws_size;
  const float* x    = (const float*)d_in[0];
  const int*   done = (const int*)d_in[1];
  const int*   act  = (const int*)d_in[2];
  const float* W1   = (const float*)d_in[3];
  const float* b1   = (const float*)d_in[4];
  const float* W2   = (const float*)d_in[5];
  const float* b2   = (const float*)d_in[6];
  const float* W3   = (const float*)d_in[7];
  const float* b3   = (const float*)d_in[8];
  const float* Wih  = (const float*)d_in[9];
  const float* Whh  = (const float*)d_in[10];
  const float* bih  = (const float*)d_in[11];
  const float* bhh  = (const float*)d_in[12];
  const float* Wlog = (const float*)d_in[13];
  const float* blog = (const float*)d_in[14];
  const float* h0   = (const float*)d_in[15];
  const float* c0   = (const float*)d_in[16];

  char* ws = (char*)d_ws;
  size_t off = 0;
  auto wsalloc = [&](size_t bytes) -> void* {
    void* p = ws + off;
    off = (off + bytes + 255) & ~(size_t)255;
    return p;
  };
  __bf16* z     = (__bf16*)wsalloc((size_t)TB * 64 * sizeof(__bf16));
  __bf16* W1b   = (__bf16*)wsalloc(128 * OBS * sizeof(__bf16));
  __bf16* W2b   = (__bf16*)wsalloc(128 * 128 * sizeof(__bf16));
  __bf16* W3b   = (__bf16*)wsalloc(64 * 128 * sizeof(__bf16));
  __bf16* Wgb   = (__bf16*)wsalloc(128 * 96 * sizeof(__bf16));
  __bf16* Wlogb = (__bf16*)wsalloc(NACT * HID * sizeof(__bf16));
  float*  biasg = (float*)wsalloc(128 * sizeof(float));

  float* outF    = (float*)d_out;
  float* outAct  = outF;                  // [T,B] action (as float)
  float* outLogp = outF + (size_t)TB;     // [T,B] logprob
  float* outEnt  = outF + (size_t)2 * TB; // [T,B] entropy

  // 0) weight prep
  prep_kernel<<<(128 * OBS + 255) / 256, 256, 0, stream>>>(
      W1, W2, W3, Wih, Whh, bih, bhh, Wlog, W1b, W2b, W3b, Wgb, biasg, Wlogb);

  // 1) encoder: padded weights + 16 waves * 2 * 16 * P2 scratch (324,608 B)
  const size_t enc_smem =
      (size_t)(128 * P1 + 128 * P2 + 64 * P2 + 16 * 2 * 16 * P2) * sizeof(__bf16);
  hipFuncSetAttribute((const void*)mlp_encoder_kernel,
                      hipFuncAttributeMaxDynamicSharedMemorySize, (int)enc_smem);
  mlp_encoder_kernel<<<256, 512, enc_smem, stream>>>(x, b1, b2, b3, W1b, W2b, W3b, z);

  // 2) LSTM scan + head: 128 waves = 128 batch tiles of 16 rows
  lstm_head_kernel<<<32, 128, 0, stream>>>(z, done, act, Wgb, biasg, Wlogb, blog,
                                           h0, c0, outLogp, outEnt);

  // 3) action pass-through
  action_copy_kernel<<<(TB + 255) / 256, 256, 0, stream>>>(act, outAct);
}
